// CoupledAttention_29240137351869
// MI455X (gfx1250) — matrix operand order repaired and checked by reading, not compiled, not run on hardware
//
#include <hip/hip_runtime.h>
#include <hip/hip_bf16.h>

// ---------------------------------------------------------------------------
// CoupledAttention on gfx1250 (MI455X).
// Flash attention on v_wmma_f32_16x16x32_f16, LDS-staged K/V^T tiles shared by
// the 4 waves of a block, precomputed (H,T,T) f16 bias (batch-invariant, lives
// in L2: 29MB << 192MB) so the softmax inner loop is 1 contiguous f16 load +
// 1 exp per score element.
// ---------------------------------------------------------------------------

typedef __attribute__((ext_vector_type(16))) _Float16 v16h;
typedef __attribute__((ext_vector_type(8)))  _Float16 v8h;
typedef __attribute__((ext_vector_type(8)))  float    v8f;

#define H_   6
#define P_   25
#define F_   64
#define T_   1600        // F_*P_
#define DIM_ 192
#define HD_  32
#define NB_  8
#define SCALE_ 0.17677669529663687f   // 1/sqrt(32)

// A-matrix (16-bit, 16x32) per-lane K index for element j (0..15):
// k = hi*8 + j (j<8) ; k = 16 + hi*8 + (j-8) (j>=8)
__device__ __forceinline__ int a_kidx(int j, int hi) {
  int v = j >> 1;
  return ((v >= 4) ? 16 : 0) + (hi << 3) + ((v & 3) << 1) + (j & 1);
}

// 16 contiguous halves -> v16h (two 16B loads)
__device__ __forceinline__ v16h ld16(const _Float16* p) {
  v8h a = *(const v8h*)p;
  v8h b = *(const v8h*)(p + 8);
  return __builtin_shufflevector(a, b, 0, 1, 2, 3, 4, 5, 6, 7,
                                 8, 9, 10, 11, 12, 13, 14, 15);
}

// A-layout fragment from a row of 32 contiguous halves
__device__ __forceinline__ v16h ld16_a(const _Float16* row, int hi) {
  v8h a = *(const v8h*)(row + (hi << 4));
  v8h b = *(const v8h*)(row + (hi << 4) + 8);
  return __builtin_shufflevector(a, b, 0, 1, 2, 3, 4, 5, 6, 7,
                                 8, 9, 10, 11, 12, 13, 14, 15);
}

// ---------------------------------------------------------------------------
// Kernel 1: qkv = x @ qkv_w^T (M=12800, N=576, K=192) -> f16 q/k/v (n,h,t,hd)
// ---------------------------------------------------------------------------
__global__ void qkv_wmma_kernel(const float* __restrict__ x,
                                const float* __restrict__ qkv_w,
                                _Float16* __restrict__ q16,
                                _Float16* __restrict__ k16,
                                _Float16* __restrict__ v16) {
  const int wave = threadIdx.x >> 5;
  const int lane = threadIdx.x & 31;
  const int hi = lane >> 4, col = lane & 15;
  const int ntiles = (3 * DIM_) / 16;            // 36
  int tile = blockIdx.x * 4 + wave;              // 800*36 = 28800 tiles
  int mt = tile / ntiles, nt = tile % ntiles;
  int gm_a = mt * 16 + col;
  int gr   = nt * 16 + col;
  v8f c = {};
  for (int kk = 0; kk < DIM_; kk += 32) {
    v16h a, b;
#pragma unroll
    for (int j = 0; j < 16; ++j) {
      a[j] = (_Float16)x[(size_t)gm_a * DIM_ + kk + (hi << 4) + j];
      b[j] = (_Float16)qkv_w[(size_t)gr * DIM_ + kk + (hi << 4) + j];
    }
    c = __builtin_amdgcn_wmma_f32_16x16x32_f16(false, a, false, b,
                                               (short)0, c, false, false);
  }
  int i3 = gr / DIM_, rr = gr % DIM_;
  int h = rr / HD_, hd = rr % HD_;
  _Float16* dst = (i3 == 0) ? q16 : ((i3 == 1) ? k16 : v16);
#pragma unroll
  for (int i = 0; i < 8; ++i) {
    int gm = mt * 16 + i + (hi << 3);
    int n = gm / T_, t = gm % T_;
    dst[(((size_t)(n * H_ + h)) * T_ + t) * HD_ + hd] = (_Float16)c[i];
  }
}

// ---------------------------------------------------------------------------
// Kernel 2: tiny bias MLP tables. biasT[h][d] (d=df+63), biasH[h][p][q].
// ---------------------------------------------------------------------------
__global__ void bias_tables_kernel(const float* __restrict__ t_w1,
                                   const float* __restrict__ t_b1,
                                   const float* __restrict__ t_w2,
                                   const float* __restrict__ t_b2,
                                   const float* __restrict__ h_w1,
                                   const float* __restrict__ h_b1,
                                   const float* __restrict__ h_w2,
                                   const float* __restrict__ h_b2,
                                   const int*   __restrict__ hop,
                                   float* __restrict__ biasT,
                                   float* __restrict__ biasH) {
  int id = blockIdx.x * blockDim.x + threadIdx.x;
  const int nT = (2 * F_ - 1) * H_;   // 762
  const int nH = P_ * P_ * H_;        // 3750
  if (id < nT) {
    int h = id / (2 * F_ - 1), d = id % (2 * F_ - 1);
    float rel = (float)(d - (F_ - 1));
    float acc = t_b2[h];
    for (int j = 0; j < DIM_; ++j) {
      float hv = rel * t_w1[j] + t_b1[j];
      if (hv > 0.f) acc += hv * t_w2[h * DIM_ + j];
    }
    biasT[h * (2 * F_ - 1) + d] = acc;
  } else if (id < nT + nH) {
    int id2 = id - nT;
    int h = id2 / (P_ * P_), pqi = id2 % (P_ * P_);
    float hf = (float)hop[pqi];
    float acc = h_b2[h];
    for (int j = 0; j < DIM_; ++j) {
      float hv = hf * h_w1[j] + h_b1[j];
      if (hv > 0.f) acc += hv * h_w2[h * DIM_ + j];
    }
    biasH[id2] = acc;   // layout h*625 + p*25 + q
  }
}

// ---------------------------------------------------------------------------
// Kernel 2b: expand to full pre-scaled bias  biasF[h][tq][tk] (f16, 29MB).
// Batch-invariant -> amortized over N=8; stays resident in the 192MB L2.
// ---------------------------------------------------------------------------
__global__ void bias_full_kernel(const float* __restrict__ biasT,
                                 const float* __restrict__ biasH,
                                 _Float16* __restrict__ biasF) {
  int idx = blockIdx.x * blockDim.x + threadIdx.x;
  const int total = H_ * T_ * T_;
  if (idx >= total) return;
  int tk = idx % T_;
  int tq = (idx / T_) % T_;
  int h  = idx / (T_ * T_);
  int fq = tq / P_, pq = tq % P_;
  int fk = tk / P_, pk = tk % P_;
  float b = biasH[(h * P_ + pq) * P_ + pk]
          + biasT[h * (2 * F_ - 1) + (fq - fk + F_ - 1)];
  biasF[idx] = (_Float16)(b * SCALE_);
}

// ---------------------------------------------------------------------------
// Kernel 3: flash attention. Block = 4 waves sharing one (n,h); per 32-key
// chunk the block stages K and V^T tiles in LDS (one global read for all 4
// waves), then each wave: 2 QK^T wmma -> bias(1 f16 ld) + online softmax ->
// P tile via per-wave LDS (C-layout -> A-layout) -> 2 PV wmma.
// ---------------------------------------------------------------------------
__global__ void flash_attn_kernel(const _Float16* __restrict__ q16,
                                  const _Float16* __restrict__ k16,
                                  const _Float16* __restrict__ v16,
                                  const _Float16* __restrict__ biasF,
                                  const float* __restrict__ alpha,
                                  float* __restrict__ ctx) {
  __shared__ _Float16 kt[32][32];       // [key][hd]
  __shared__ _Float16 vtT[32][32];      // [hd][key]  (transposed V tile)
  __shared__ _Float16 pt[4][16][32];    // per-wave probability tile
  const int tid  = threadIdx.x;
  const int wave = tid >> 5;
  const int lane = tid & 31;
  const int hi = lane >> 4, col = lane & 15;
  const int nh = blockIdx.x / 25;                  // 0..47 = n*H+h
  const int h  = nh % H_;
  const int n  = nh / H_;
  const int qt = (blockIdx.x % 25) * 4 + wave;     // 0..99
  const size_t base = (size_t)nh * T_ * HD_;

  // Q fragment (A-layout), two 16B global loads
  const v16h aq = ld16_a(q16 + base + (size_t)(qt * 16 + col) * HD_, hi);

  // per-row bias row pointers (row of acc VGPR i is qt*16 + i + 8*hi)
  const _Float16* brow[8];
#pragma unroll
  for (int i = 0; i < 8; ++i)
    brow[i] = biasF + ((size_t)h * T_ + (qt * 16 + i + (hi << 3))) * T_;

  float rmax[8], rsum[8];
  v8f o0 = {}, o1 = {};
#pragma unroll
  for (int i = 0; i < 8; ++i) { rmax[i] = -1e30f; rsum[i] = 0.f; }

  const int r  = tid >> 2;          // 0..31 staging row (key)
  const int cs = (tid & 3) << 3;    // 0,8,16,24 staging hd chunk

  for (int kb = 0; kb < T_; kb += 32) {
    // ---- cooperative staging: K tile + transposed V tile ----
    v8h kv = *(const v8h*)&k16[base + (size_t)(kb + r) * HD_ + cs];
    v8h vv = *(const v8h*)&v16[base + (size_t)(kb + r) * HD_ + cs];
    *(v8h*)&kt[r][cs] = kv;
#pragma unroll
    for (int m = 0; m < 8; ++m) vtT[cs + m][r] = vv[m];
    __syncthreads();

    // ---- scores: 2x QK^T wmma from LDS ----
    v8f s0 = {}, s1 = {};
    v16h bk = ld16(&kt[col][hi << 4]);
    s0 = __builtin_amdgcn_wmma_f32_16x16x32_f16(false, aq, false, bk,
                                                (short)0, s0, false, false);
    bk = ld16(&kt[16 + col][hi << 4]);
    s1 = __builtin_amdgcn_wmma_f32_16x16x32_f16(false, aq, false, bk,
                                                (short)0, s1, false, false);

    // ---- online softmax (rows span the 16 lanes of one EXEC half) ----
#pragma unroll
    for (int i = 0; i < 8; ++i) {
      float e0 = s0[i] * SCALE_ + (float)brow[i][kb + col];
      float e1 = s1[i] * SCALE_ + (float)brow[i][kb + 16 + col];
      float m = fmaxf(e0, e1);
#pragma unroll
      for (int d = 1; d < 16; d <<= 1) m = fmaxf(m, __shfl_xor(m, d, 32));
      float nm  = fmaxf(rmax[i], m);
      float fac = __expf(rmax[i] - nm);
      float p0  = __expf(e0 - nm);
      float p1  = __expf(e1 - nm);
      float ps  = p0 + p1;
#pragma unroll
      for (int d = 1; d < 16; d <<= 1) ps += __shfl_xor(ps, d, 32);
      rsum[i] = rsum[i] * fac + ps;
      o0[i] *= fac; o1[i] *= fac;
      rmax[i] = nm;
      pt[wave][i + (hi << 3)][col]      = (_Float16)p0;
      pt[wave][i + (hi << 3)][16 + col] = (_Float16)p1;
    }
    __syncthreads();   // fence P write -> cross-lane read

    // ---- PV: P (A-layout) x V^T tile -> 2 wmma ----
    v16h ap = ld16_a(&pt[wave][col][0], hi);
    v16h bv = ld16(&vtT[col][hi << 4]);
    o0 = __builtin_amdgcn_wmma_f32_16x16x32_f16(false, ap, false, bv,
                                                (short)0, o0, false, false);
    bv = ld16(&vtT[16 + col][hi << 4]);
    o1 = __builtin_amdgcn_wmma_f32_16x16x32_f16(false, ap, false, bv,
                                                (short)0, o1, false, false);
    __syncthreads();   // protect next iteration's staging (WAR)
  }

  const float al = alpha[0];
#pragma unroll
  for (int i = 0; i < 8; ++i) {
    int tq = qt * 16 + i + (hi << 3);
    float inv = al / rsum[i];
    size_t ob = ((size_t)n * T_ + tq) * DIM_ + h * HD_;
    ctx[ob + col]      = o0[i] * inv;
    ctx[ob + 16 + col] = o1[i] * inv;
  }
}

// ---------------------------------------------------------------------------
// Kernel 4: relational block-diagonal term; ctx += einsum('hpq,nhfqc', ...)
// ---------------------------------------------------------------------------
__global__ void rel_ctx_kernel(const _Float16* __restrict__ v16,
                               const float* __restrict__ outer,
                               float* __restrict__ ctx) {
  int idx = blockIdx.x * blockDim.x + threadIdx.x;
  const int total = NB_ * H_ * F_ * P_ * HD_;
  if (idx >= total) return;
  int hd = idx % HD_;
  int p  = (idx / HD_) % P_;
  int f  = (idx / (HD_ * P_)) % F_;
  int h  = (idx / (HD_ * P_ * F_)) % H_;
  int n  = idx / (HD_ * P_ * F_ * H_);
  size_t vb = ((size_t)(n * H_ + h) * T_ + f * P_) * HD_ + hd;
  float acc = 0.f;
#pragma unroll
  for (int q = 0; q < P_; ++q)
    acc += outer[(h * P_ + p) * P_ + q] * (float)v16[vb + (size_t)q * HD_];
  ctx[((size_t)n * T_ + f * P_ + p) * DIM_ + h * HD_ + hd] += acc;
}

// ---------------------------------------------------------------------------
// Kernel 5: out = ctx @ proj_w^T + proj_b  (M=12800, N=192, K=192), WMMA.
// ---------------------------------------------------------------------------
__global__ void proj_wmma_kernel(const float* __restrict__ ctx,
                                 const float* __restrict__ proj_w,
                                 const float* __restrict__ proj_b,
                                 float* __restrict__ out) {
  const int wave = threadIdx.x >> 5;
  const int lane = threadIdx.x & 31;
  const int hi = lane >> 4, col = lane & 15;
  const int ntiles = DIM_ / 16;                 // 12
  int tile = blockIdx.x * 4 + wave;             // 9600 tiles
  int mt = tile / ntiles, nt = tile % ntiles;
  int gm_a = mt * 16 + col;
  int gr   = nt * 16 + col;
  v8f c = {};
  for (int kk = 0; kk < DIM_; kk += 32) {
    v16h a, b;
#pragma unroll
    for (int j = 0; j < 16; ++j) {
      a[j] = (_Float16)ctx[(size_t)gm_a * DIM_ + kk + (hi << 4) + j];
      b[j] = (_Float16)proj_w[(size_t)gr * DIM_ + kk + (hi << 4) + j];
    }
    c = __builtin_amdgcn_wmma_f32_16x16x32_f16(false, a, false, b,
                                               (short)0, c, false, false);
  }
  float bb = proj_b[gr];
#pragma unroll
  for (int i = 0; i < 8; ++i) {
    int gm = mt * 16 + i + (hi << 3);
    out[(size_t)gm * DIM_ + gr] = c[i] + bb;
  }
}

// ---------------------------------------------------------------------------
extern "C" void kernel_launch(void* const* d_in, const int* in_sizes, int n_in,
                              void* d_out, int out_size, void* d_ws, size_t ws_size,
                              hipStream_t stream) {
  const float* x      = (const float*)d_in[0];
  const float* qkv_w  = (const float*)d_in[1];
  const float* proj_w = (const float*)d_in[2];
  const float* proj_b = (const float*)d_in[3];
  const float* t_w1   = (const float*)d_in[4];
  const float* t_b1   = (const float*)d_in[5];
  const float* t_w2   = (const float*)d_in[6];
  const float* t_b2   = (const float*)d_in[7];
  const float* h_w1   = (const float*)d_in[8];
  const float* h_b1   = (const float*)d_in[9];
  const float* h_w2   = (const float*)d_in[10];
  const float* h_b2   = (const float*)d_in[11];
  const float* outer  = (const float*)d_in[12];
  const float* alpha  = (const float*)d_in[13];
  const int*   hop    = (const int*)d_in[14];
  float* out = (float*)d_out;

  // workspace carve-up (~54 MB)
  char* ws = (char*)d_ws;
  const size_t nqkv = (size_t)NB_ * H_ * T_ * HD_;        // 2,457,600 elems
  _Float16* q16 = (_Float16*)ws;        ws += nqkv * sizeof(_Float16);
  _Float16* k16 = (_Float16*)ws;        ws += nqkv * sizeof(_Float16);
  _Float16* v16 = (_Float16*)ws;        ws += nqkv * sizeof(_Float16);
  float* biasT  = (float*)ws;           ws += 4096;       // 762 floats
  float* biasH  = (float*)ws;           ws += 16384;      // 3750 floats
  _Float16* biasF = (_Float16*)ws;      ws += (size_t)H_ * T_ * T_ * sizeof(_Float16);
  float* ctx    = (float*)ws;           // 12800*192 floats

  // 1) qkv projection (WMMA)
  qkv_wmma_kernel<<<7200, 128, 0, stream>>>(x, qkv_w, q16, k16, v16);
  // 2) bias MLP tables, then full pre-scaled (H,T,T) f16 bias
  bias_tables_kernel<<<36, 128, 0, stream>>>(t_w1, t_b1, t_w2, t_b2,
                                             h_w1, h_b1, h_w2, h_b2,
                                             hop, biasT, biasH);
  bias_full_kernel<<<(H_ * T_ * T_ + 255) / 256, 256, 0, stream>>>(biasT, biasH,
                                                                   biasF);
  // 3) flash attention (WMMA, LDS-staged K/V^T)
  flash_attn_kernel<<<1200, 128, 0, stream>>>(q16, k16, v16, biasF, alpha, ctx);
  // 4) relational block-diagonal add
  rel_ctx_kernel<<<9600, 256, 0, stream>>>(v16, outer, ctx);
  // 5) output projection (WMMA)
  proj_wmma_kernel<<<2400, 128, 0, stream>>>(ctx, proj_w, proj_b, out);
}
